// MultiHeadAttention_18193481465933
// MI455X (gfx1250) — matrix operand order repaired and checked by reading, not compiled, not run on hardware
//
#include <hip/hip_runtime.h>
#include <hip/hip_bf16.h>

// ---------------------------------------------------------------------------
// MHA forward for MI455X (gfx1250, wave32).
//   out   = (x@W.T+b projections) -> per-head causal softmax(QK^T/8)@V -> @Wo.T+bo
//   d_out = [ out : 8192*1024 f32 | attn_weights : 4*16*2048*2048 f32 ]
// Precision: bf16 A/B fragments, f32 accumulation (v_wmma_f32_16x16x32_bf16).
// Roofline: attn_weights store (1.07 GB) dominates -> ~46 us @ 23.3 TB/s;
// bf16 WMMA keeps the 137 GFLOP of GEMMs well under that floor. Q/K/V (48 MB
// bf16) stay L2-resident; prob stream uses non-temporal stores.
// Workspace: Qh,Kh,Vt,merged bf16 = 64 MiB.
// ---------------------------------------------------------------------------

typedef __bf16 bf16;
typedef __attribute__((ext_vector_type(16))) __bf16 v16bf;
typedef __attribute__((ext_vector_type(8)))  float  v8f;
typedef __attribute__((ext_vector_type(4)))  float  v4f;

#define EMBED 1024
#define NHEAD 16
#define HDIM  64
#define BATCH 4
#define SEQ   2048
#define NTOK  (BATCH * SEQ)   // 8192
#define LDP   (SEQ + 4)       // padded LDS leading dim (bank decorrelation)

static __device__ __forceinline__ int lane_id() { return (int)(threadIdx.x & 31); }

static __device__ __forceinline__ v8f vzero8() {
  v8f z = {0.f, 0.f, 0.f, 0.f, 0.f, 0.f, 0.f, 0.f};
  return z;
}

// ---- WMMA fragment loaders (contiguous k = 16h+i in A and B, not the ISA table order) ------
// A fragment 16x32 (MxK) bf16: lane l (m=l&15, h=l>>4) holds one contiguous
// 16-element chunk at k = 16h .. 16h+15 of row m.
static __device__ __forceinline__ v16bf load_a_bf16(const bf16* src, int ld) {
  int l = lane_id(), m = l & 15, h = l >> 4;
  const bf16* p = src + (size_t)m * ld + 16 * h;
  v16bf a;
#pragma unroll
  for (int i = 0; i < 16; ++i) a[i] = p[i];
  return a;
}

static __device__ __forceinline__ v16bf load_a_f32(const float* src, int ld) {
  int l = lane_id(), m = l & 15, h = l >> 4;
  const float* p = src + (size_t)m * ld + 16 * h;
  v16bf a;
#pragma unroll
  for (int i = 0; i < 16; ++i) a[i] = (bf16)p[i];
  return a;
}

// B fragment 32x16 (KxN), column n contiguous along k in memory (src [N][K]):
// lane l (n=l&15, h=l>>4) holds one 16-element chunk at k = 16h.
static __device__ __forceinline__ v16bf load_b_tr_f32(const float* src, int ld) {
  int l = lane_id(), n = l & 15, h = l >> 4;
  const float* p = src + (size_t)n * ld + 16 * h;
  v16bf b;
#pragma unroll
  for (int i = 0; i < 16; ++i) b[i] = (bf16)p[i];
  return b;
}

static __device__ __forceinline__ v16bf load_b_tr_bf16(const bf16* src, int ld) {
  int l = lane_id(), n = l & 15, h = l >> 4;
  const bf16* p = src + (size_t)n * ld + 16 * h;
  v16bf b;
#pragma unroll
  for (int i = 0; i < 16; ++i) b[i] = p[i];
  return b;
}

static __device__ __forceinline__ v8f wmma_bf16(v16bf a, v16bf b, v8f c) {
  return __builtin_amdgcn_wmma_f32_16x16x32_bf16(
      /*neg_a=*/false, a, /*neg_b=*/false, b,
      /*c_mod=*/(short)0, c, /*reuse_a=*/false, /*reuse_b=*/false);
}

// ---------------------------------------------------------------------------
// Projection: out = X @ W^T + bias ; X [8192,1024] f32, W [1024,1024] f32.
// VTRANS=0: head-split [B,H,S,Dh] (Q,K). VTRANS=1: transposed [B,H,Dh,S] (V),
// which makes the P@V B-fragments contiguous 32B-per-lane loads.
// Workgroup = 128 threads (4 waves); WG tile 64x64; wave tile 16x64.
// ---------------------------------------------------------------------------
template <bool VTRANS>
__global__ __launch_bounds__(128) void proj_qkv_kernel(
    const float* __restrict__ X, const float* __restrict__ W,
    const float* __restrict__ bias, bf16* __restrict__ outHS) {
  int wave = (int)(threadIdx.x >> 5);
  int tm = (int)blockIdx.x * 64 + wave * 16;  // row (token) tile
  int tn = (int)blockIdx.y * 64;              // col (embed) tile

  v8f c[4];
#pragma unroll
  for (int j = 0; j < 4; ++j) c[j] = vzero8();

  for (int kk = 0; kk < EMBED; kk += 32) {
    v16bf a = load_a_f32(X + (size_t)tm * EMBED + kk, EMBED);
#pragma unroll
    for (int j = 0; j < 4; ++j) {
      v16bf bw = load_b_tr_f32(W + (size_t)(tn + 16 * j) * EMBED + kk, EMBED);
      c[j] = wmma_bf16(a, bw, c[j]);
    }
  }

  // C layout: value (r,l) -> row = r + 8*(l>=16), col = l&15.
  int l = lane_id(), n = l & 15, hc = l >> 4;
#pragma unroll
  for (int j = 0; j < 4; ++j) {
    int e = tn + 16 * j + n;
    float bv = bias[e];
    int hh = e >> 6, d = e & 63;
#pragma unroll
    for (int r = 0; r < 8; ++r) {
      int row = tm + r + 8 * hc;          // global token index
      int bb = row >> 11, s = row & (SEQ - 1);
      size_t idx;
      if (VTRANS)
        idx = (((size_t)bb * NHEAD + hh) * HDIM + d) * SEQ + s;   // [B,H,D,S]
      else
        idx = (((size_t)bb * NHEAD + hh) * SEQ + s) * HDIM + d;   // [B,H,S,D]
      outHS[idx] = (bf16)(c[j][r] + bv);
    }
  }
}

// ---------------------------------------------------------------------------
// Output projection: out = A @ Wo^T + bo ; A bf16 [8192,1024], out f32 d_out.
// ---------------------------------------------------------------------------
__global__ __launch_bounds__(128) void proj_out_kernel(
    const bf16* __restrict__ A, const float* __restrict__ W,
    const float* __restrict__ bias, float* __restrict__ out) {
  int wave = (int)(threadIdx.x >> 5);
  int tm = (int)blockIdx.x * 64 + wave * 16;
  int tn = (int)blockIdx.y * 64;

  v8f c[4];
#pragma unroll
  for (int j = 0; j < 4; ++j) c[j] = vzero8();

  for (int kk = 0; kk < EMBED; kk += 32) {
    v16bf a = load_a_bf16(A + (size_t)tm * EMBED + kk, EMBED);
#pragma unroll
    for (int j = 0; j < 4; ++j) {
      v16bf bw = load_b_tr_f32(W + (size_t)(tn + 16 * j) * EMBED + kk, EMBED);
      c[j] = wmma_bf16(a, bw, c[j]);
    }
  }

  int l = lane_id(), n = l & 15, hc = l >> 4;
#pragma unroll
  for (int j = 0; j < 4; ++j) {
    int e = tn + 16 * j + n;
    float bv = bias[e];
#pragma unroll
    for (int r = 0; r < 8; ++r) {
      int row = tm + r + 8 * hc;
      __builtin_nontemporal_store(c[j][r] + bv, out + (size_t)row * EMBED + e);
    }
  }
}

// ---------------------------------------------------------------------------
// Causal attention for one (b, h, 16-query-row tile). 256 threads = 8 waves.
// LDS: 16 x LDP f32 score tile (~128 KiB) + 16x64 f32 reduce + 2 KiB Q tile.
// Phase 0: Q tile staged to LDS via global_load_async_to_lds_b128 (ASYNCcnt).
// Phase 1: scores = (Q K^T)/8 up to the diagonal, via WMMA -> LDS.
// Phase 2: float4 softmax (16-lane shuffle reductions), non-temporal prob +
//          zero stores to d_out.
// Phase 3: O = P @ V (V transposed layout), all 8 waves split K, LDS reduce.
// ---------------------------------------------------------------------------
__global__ __launch_bounds__(256) void attn_kernel(
    const bf16* __restrict__ Q, const bf16* __restrict__ K,
    const bf16* __restrict__ Vt, float* __restrict__ attnW,
    bf16* __restrict__ merged) {
  __shared__ float sP[16 * LDP];
  __shared__ float sO[16 * HDIM];
  __shared__ __align__(16) bf16 sQ[16 * HDIM];  // 2 KiB

  int qt = (int)blockIdx.x;       // query tile index (16 rows)
  int h  = (int)blockIdx.y;
  int b  = (int)blockIdx.z;
  int qrow0 = qt * 16;
  size_t bh = (size_t)(b * NHEAD + h);

  const bf16* Qp  = Q  + (bh * SEQ + (size_t)qrow0) * HDIM;
  const bf16* Kp  = K  + bh * SEQ * HDIM;
  const bf16* Vtp = Vt + bh * HDIM * SEQ;       // [Dh][S] per head
  float* Ap = attnW + (bh * SEQ + (size_t)qrow0) * SEQ;

  int wave = (int)(threadIdx.x >> 5);
  int l = lane_id(), n = l & 15, hc = l >> 4;

  // ---- Phase 0: async-copy the 16x64 bf16 Q tile (2 KiB) into LDS ---------
  {
    int t = (int)threadIdx.x;
    if (t < 128) {
      unsigned ldso = (unsigned)(size_t)(&sQ[0]) + (unsigned)(t * 16);
      const char* g = (const char*)Qp + t * 16;
      asm volatile("global_load_async_to_lds_b128 %0, %1, off"
                   :: "v"(ldso), "v"(g) : "memory");
    }
    asm volatile("s_wait_asynccnt 0" ::: "memory");
  }
  __syncthreads();

  // ---- Phase 1: score tiles up to (and including) the diagonal tile -------
  v16bf aQ0 = load_a_bf16(sQ, HDIM);       // d = 0..31  (from LDS)
  v16bf aQ1 = load_a_bf16(sQ + 32, HDIM);  // d = 32..63
  int nkt = qt + 1;                        // key tiles with any unmasked entry
  for (int jt = wave; jt < nkt; jt += 8) {
    int key0 = jt * 16;
    if (jt + 8 < nkt)
      __builtin_prefetch(Kp + (size_t)(key0 + 128) * HDIM, 0, 1);
    v16bf b0 = load_b_tr_bf16(Kp + (size_t)key0 * HDIM, HDIM);
    v16bf b1 = load_b_tr_bf16(Kp + (size_t)key0 * HDIM + 32, HDIM);
    v8f c = vzero8();
    c = wmma_bf16(aQ0, b0, c);
    c = wmma_bf16(aQ1, b1, c);
#pragma unroll
    for (int r = 0; r < 8; ++r) {
      int m = r + 8 * hc;
      float s = c[r] * 0.125f;                       // 1/sqrt(64)
      if (key0 + n > qrow0 + m) s = -1e30f;          // causal mask
      sP[m * LDP + key0 + n] = s;
    }
  }
  __syncthreads();

  // ---- Phase 2: row softmax + non-temporal global write of probabilities --
  {
    int t = (int)threadIdx.x;
    int row = t >> 4, lc = t & 15;     // 16 threads per row
    int ncols = qrow0 + 16;            // valid columns (multiple of 16)
    int nc4 = ncols >> 2;
    v4f* pr4 = (v4f*)(sP + row * LDP);

    float mx = -1e30f;
    for (int c4 = lc; c4 < nc4; c4 += 16) {
      v4f x = pr4[c4];
      mx = fmaxf(mx, fmaxf(fmaxf(x[0], x[1]), fmaxf(x[2], x[3])));
    }
#pragma unroll
    for (int off = 8; off >= 1; off >>= 1)
      mx = fmaxf(mx, __shfl_xor(mx, off, 32));

    float sum = 0.f;
    for (int c4 = lc; c4 < nc4; c4 += 16) {
      v4f x = pr4[c4], e;
#pragma unroll
      for (int i = 0; i < 4; ++i) e[i] = __expf(x[i] - mx);
      pr4[c4] = e;
      sum += (e[0] + e[1]) + (e[2] + e[3]);
    }
#pragma unroll
    for (int off = 8; off >= 1; off >>= 1) sum += __shfl_xor(sum, off, 32);
    float inv = 1.0f / sum;

    v4f* go4 = (v4f*)(Ap + (size_t)row * SEQ);
    for (int c4 = lc; c4 < nc4; c4 += 16) {
      v4f p = pr4[c4] * inv;
      pr4[c4] = p;                                   // kept for P@V
      __builtin_nontemporal_store(p, go4 + c4);      // attn_weights output
    }
    v4f z4 = {0.f, 0.f, 0.f, 0.f};
    // pad LDS to a multiple of 32 columns for the phase-3 K loop
    if ((ncols & 31) && lc < 4) pr4[nc4 + lc] = z4;
    // masked region of attn_weights is exactly zero
    for (int c4 = nc4 + lc; c4 < SEQ / 4; c4 += 16)
      __builtin_nontemporal_store(z4, go4 + c4);
  }
  __syncthreads();

  // ---- Phase 3: O = P @ V ; 8 waves: (wave&3) picks Dh slice, wave>>2 the
  // interleaved K half; partials reduced through LDS. ------------------------
  {
    int g = wave >> 2;                 // 0 or 1: K interleave group
    int n0 = (wave & 3) * 16;          // 16-wide Dh slice
    v8f c = vzero8();
    int kend = qrow0 + 16;
    for (int kk = g * 32; kk < kend; kk += 64) {
      __builtin_prefetch(Vtp + (size_t)n0 * SEQ + kk + 64, 0, 1);
      v16bf a  = load_a_f32(sP + kk, LDP);                    // P (LDS)
      v16bf bv = load_b_tr_bf16(Vtp + (size_t)n0 * SEQ + kk, SEQ);
      c = wmma_bf16(a, bv, c);
    }
    if (g == 1) {
#pragma unroll
      for (int r = 0; r < 8; ++r)
        sO[(r + 8 * hc) * HDIM + n0 + n] = c[r];
    }
    __syncthreads();
    if (g == 0) {
#pragma unroll
      for (int r = 0; r < 8; ++r) {
        int row = r + 8 * hc;
        float o = c[r] + sO[row * HDIM + n0 + n];
        int sg = qrow0 + row;
        int e = h * HDIM + n0 + n;
        merged[((size_t)b * SEQ + sg) * EMBED + e] = (bf16)o;
      }
    }
  }
}

// ---------------------------------------------------------------------------
extern "C" void kernel_launch(void* const* d_in, const int* in_sizes, int n_in,
                              void* d_out, int out_size, void* d_ws,
                              size_t ws_size, hipStream_t stream) {
  const float* q  = (const float*)d_in[0];
  const float* k  = (const float*)d_in[1];
  const float* v  = (const float*)d_in[2];
  // d_in[3] = causal mask (known analytically; unused)
  const float* Wq = (const float*)d_in[4];
  const float* bq = (const float*)d_in[5];
  const float* Wk = (const float*)d_in[6];
  const float* bk = (const float*)d_in[7];
  const float* Wv = (const float*)d_in[8];
  const float* bv = (const float*)d_in[9];
  const float* Wo = (const float*)d_in[10];
  const float* bo = (const float*)d_in[11];

  const size_t nQKV = (size_t)BATCH * NHEAD * SEQ * HDIM;  // 8,388,608
  bf16* wsQ = (bf16*)d_ws;
  bf16* wsK = wsQ + nQKV;
  bf16* wsV = wsK + nQKV;   // V in transposed [B,H,Dh,S] layout
  bf16* wsM = wsV + nQKV;   // merged attention output [B,S,E] bf16

  float* outO = (float*)d_out;                       // [B,S,E] f32
  float* outA = outO + (size_t)NTOK * EMBED;         // [B,H,S,S] f32

  dim3 gG(NTOK / 64, EMBED / 64);   // 128 x 16 workgroups
  dim3 bG(128);
  proj_qkv_kernel<false><<<gG, bG, 0, stream>>>(q, Wq, bq, wsQ);
  proj_qkv_kernel<false><<<gG, bG, 0, stream>>>(k, Wk, bk, wsK);
  proj_qkv_kernel<true ><<<gG, bG, 0, stream>>>(v, Wv, bv, wsV);

  dim3 gA(SEQ / 16, NHEAD, BATCH);  // 128 x 16 x 4
  attn_kernel<<<gA, dim3(256), 0, stream>>>(wsQ, wsK, wsV, outA, wsM);

  proj_out_kernel<<<gG, bG, 0, stream>>>(wsM, Wo, bo, outO);
}
